// LASNet_60275571032828
// MI455X (gfx1250) — matrix operand order repaired and checked by reading, not compiled, not run on hardware
//
#include <hip/hip_runtime.h>

// ---------------------------------------------------------------------------
// LASNet forward on MI455X (gfx1250): bf16 WMMA implicit-GEMM 3x3 convs with
// double-buffered LDS + TDM (tensor_load_to_lds) weight staging, fused
// mix/argmax/select, fused dual-ConvLSTM gates. NHWC internal layout.
// ---------------------------------------------------------------------------

#define BB 2
#define TT 5
#define CC 32
#define HH 96
#define WW 96

typedef __attribute__((ext_vector_type(16))) __bf16 v16bf;
typedef __attribute__((ext_vector_type(8)))  __bf16 v8bf;
typedef __attribute__((ext_vector_type(4)))  __bf16 v4bf;
typedef __attribute__((ext_vector_type(8)))  float  v8f;
typedef __attribute__((ext_vector_type(4)))  unsigned int v4u;
typedef __attribute__((ext_vector_type(8)))  int    v8i;
typedef __attribute__((ext_vector_type(4)))  int    v4i;

static constexpr int SLICE = BB * HH * WW * CC;   // 589824 elems, one [B,H,W,32] slice
static constexpr int NPIX  = BB * HH * WW;        // 18432 pixels

#if defined(__has_builtin)
#  if __has_builtin(__builtin_amdgcn_tensor_load_to_lds)
#    define HAVE_TDM 1
#  else
#    define HAVE_TDM 0
#  endif
#  if __has_builtin(__builtin_amdgcn_s_wait_tensorcnt)
#    define WAIT_TENSOR(n) __builtin_amdgcn_s_wait_tensorcnt((short)(n))
#  else
#    define WAIT_TENSOR(n)
#  endif
#else
#  define HAVE_TDM 0
#  define WAIT_TENSOR(n)
#endif

__device__ __forceinline__ float sigf(float x) { return 1.0f / (1.0f + __expf(-x)); }

// ------------------------------- utility kernels ---------------------------

__global__ void k_zero(float* __restrict__ p, int n) {
    int i = blockIdx.x * 256 + threadIdx.x;
    if (i < n) p[i] = 0.0f;
}

// x[B,T,C,H,W] fp32 -> dst[t][b][y][x][c] fp32
__global__ void k_transpose_in(const float* __restrict__ x, float* __restrict__ dst, int n) {
    int i = blockIdx.x * 256 + threadIdx.x;
    if (i >= n) return;
    int c  = i & (CC - 1);
    int xx = (i / CC) % WW;
    int y  = (i / (CC * WW)) % HH;
    int b  = (i / (CC * WW * HH)) % BB;
    int t  =  i / (CC * WW * HH * BB);
    dst[i] = x[((((size_t)b * TT + t) * CC + c) * HH + y) * WW + xx];
}

// cur[t][b][y][x][c] -> d_out[b][t][c][y][x]
__global__ void k_unpack(const float* __restrict__ src, float* __restrict__ dst, int n) {
    int i = blockIdx.x * 256 + threadIdx.x;
    if (i >= n) return;
    int xx = i % WW;
    int y  = (i / WW) % HH;
    int c  = (i / (WW * HH)) % CC;
    int t  = (i / (WW * HH * CC)) % TT;
    int b  =  i / (WW * HH * CC * TT);
    dst[i] = src[((((size_t)t * BB + b) * HH + y) * WW + xx) * CC + c];
}

// w[O=128][Cin][3][3] fp32 -> dst[n][k] bf16, k = (ky*3+kx)*Cin + c
__global__ void k_wpack(const float* __restrict__ w, __bf16* __restrict__ dst, int Cin, int n) {
    int i = blockIdx.x * 256 + threadIdx.x;
    if (i >= n) return;
    int K9 = 9 * Cin;
    int no = i / K9;
    int k  = i - no * K9;
    int p  = k / Cin;
    int c  = k - p * Cin;
    dst[i] = (__bf16)w[((size_t)no * Cin + c) * 9 + p];
}

// _select(_mix_t(...)): softmax is monotone => argmax over the pre-softmax
// logits at flat-offset 0 of each slice; then emit only the selected slice.
// Slices s < zp are implicit zeros; slices s >= zp live at src + (s-zp)*ss.
__global__ void k_mix_select(float* __restrict__ out, const float* __restrict__ src,
                             const float* __restrict__ w, const float* __restrict__ bias,
                             int S, int zp, long long ss, int n) {
    int i = blockIdx.x * 256 + threadIdx.x;
    if (i >= n) return;
    int idx = 0; float best = -3.402823e38f;
    for (int sp = 0; sp < S; ++sp) {
        float v = bias[sp];
        for (int s = zp; s < S; ++s) v += w[sp * S + s] * src[(long long)(s - zp) * ss];
        if (v > best) { best = v; idx = sp; }
    }
    float acc = bias[idx];
    for (int s = zp; s < S; ++s) acc += w[idx * S + s] * src[(long long)(s - zp) * ss + i];
    out[i] = acc;
}

// channel-concat + fp32->bf16 packs (NHWC fast-channel)
__global__ void k_pack2(__bf16* __restrict__ dst, const float* __restrict__ a,
                        const float* __restrict__ b, int n) {
    int i = blockIdx.x * 256 + threadIdx.x;
    if (i >= n) return;
    int c = i & 63, pix = i >> 6;
    float v = (c < 32) ? a[(size_t)pix * 32 + c] : b[(size_t)pix * 32 + (c - 32)];
    dst[i] = (__bf16)v;
}

__global__ void k_pack4(__bf16* __restrict__ dst, const float* __restrict__ a,
                        const float* __restrict__ b, const float* __restrict__ c2,
                        const float* __restrict__ d2, int n) {
    int i = blockIdx.x * 256 + threadIdx.x;
    if (i >= n) return;
    int c = i & 127, pix = i >> 7;
    int g = c >> 5, cl = c & 31;
    const float* s = (g == 0) ? a : (g == 1) ? b : (g == 2) ? c2 : d2;
    dst[i] = (__bf16)s[(size_t)pix * 32 + cl];
}

// fused dual gated ConvLSTM update (shared gate logits)
__global__ void k_gates(const float* __restrict__ cc, const float* __restrict__ mm,
                        const float* __restrict__ cprev, const float* __restrict__ mcprev,
                        float* __restrict__ hN, float* __restrict__ cN,
                        float* __restrict__ mhN, float* __restrict__ mcN,
                        float* __restrict__ outs, int n) {
    int i = blockIdx.x * 256 + threadIdx.x;
    if (i >= n) return;
    int c = i & 31, pix = i >> 5;
    size_t b = (size_t)pix * 128 + c;
    float ci = cc[b], cf = cc[b + 32], co = cc[b + 64], cg = cc[b + 96];
    float cn = sigf(cf) * cprev[i] + sigf(ci) * tanhf(cg);
    float hn = sigf(co) * tanhf(cn);
    float mi = mm[b] + ci, mf = mm[b + 32] + cf, mo = mm[b + 64] + co, mg = mm[b + 96] + cg;
    float mcn = sigf(mf) * mcprev[i] + sigf(mi) * tanhf(mg);
    float mhn = sigf(mo) * tanhf(mcn);
    hN[i] = hn; cN[i] = cn; mhN[i] = mhn; mcN[i] = mcn; outs[i] = hn + mhn;
}

// ------------------------- WMMA implicit-GEMM 3x3 conv ---------------------
// in : [B*H*W, CTOT] bf16 (NHWC)    wp : [128, 9*CTOT] bf16 ([N][K], K contig)
// out: [B*H*W, 128] fp32 (+bias)
// Block = 256 thr (8 waves) -> 32 pixels x 128 outputs.
// Double-buffered LDS: stage K-step k+1 (A: vector copies; B: one TDM 2D
// descriptor with LDS padding -> 40-elem rows) while WMMAs consume K-step k.
// TDM completion ordered by s_wait_tensorcnt(<=1 outstanding) + barrier.
// Fragment M/N/acc mapping per CDNA5 ISA 7.12.2; K order contiguous (khalf*16 + e) in A and B.
template <int CTOT>
__global__ __launch_bounds__(256) void k_conv(const __bf16* __restrict__ in,
                                              const __bf16* __restrict__ wp,
                                              const float* __restrict__ bias,
                                              float* __restrict__ out) {
    constexpr int CHUNKS = CTOT / 32;
    constexpr int KSTEPS = 9 * CHUNKS;
    constexpr int KROW   = 9 * CTOT;
    __shared__ __bf16 As[2][32][40];    // 32 pixels x 32 K, 80B rows (16B-aligned)
    __shared__ __bf16 Bs[2][128][40];   // [N][K] so B-fragment K-run is contiguous

    const int tid    = threadIdx.x;
    const int lane   = tid & 31;
    const int wv     = tid >> 5;
    const int lrow   = lane & 15;
    const int khalf  = lane >> 4;
    const int msub   = wv >> 2;
    const int ntile0 = (wv & 3) * 2;
    const int mbase  = blockIdx.x * 32;     // W%32==0: tile never crosses a row

    // A staging: 1024 bf16 / 256 thr = 4 contiguous channels each (b64)
    const int a_m  = tid >> 3;
    const int a_c4 = (tid & 7) * 4;
    const int a_gm = mbase + a_m;
    const int a_b  = a_gm / (HH * WW);
    const int a_r  = a_gm - a_b * (HH * WW);
    const int a_y  = a_r / WW;
    const int a_x  = a_r - a_y * WW;

#if !HAVE_TDM
    const int b_n  = tid >> 1;              // fallback manual B staging
    const int b_k0 = (tid & 1) * 16;
#endif

    auto stageA = [&](int ks, int buf) {
        const int p  = ks / CHUNKS;
        const int cb = ks - p * CHUNKS;
        const int yy = a_y + (p / 3) - 1;
        const int xx = a_x + (p - (p / 3) * 3) - 1;
        v4bf av = {};
        if (((unsigned)yy < HH) && ((unsigned)xx < WW)) {
            const __bf16* s = in + ((size_t)(a_b * HH + yy) * WW + xx) * CTOT + cb * 32 + a_c4;
            av = *(const v4bf*)s;
            if (cb + 1 < CHUNKS) __builtin_prefetch((const void*)(s + 32), 0, 1);
        }
        *(v4bf*)&As[buf][a_m][a_c4] = av;
    };

    auto stageB = [&](int ks, int buf) {
#if HAVE_TDM
        if (wv != 0) return;   // one wave drives the DMA engine
        const unsigned long long ga = (unsigned long long)(size_t)wp + (size_t)ks * 64ull;
        const unsigned lds = (unsigned)(size_t)&Bs[buf][0][0];
        // D# group0: count=1 | lds_addr | global_addr(57b) | type=2
        v4u g0 = { 1u, lds, (unsigned)ga,
                   (unsigned)((ga >> 32) & 0x01FFFFFFu) | (2u << 30) };
        // D# group1: data_size=2B, pad_enable, pad_interval=16DW(64B=row),
        // pad_amount=4DW(16B -> 40-elem LDS rows); tensor 2D [KROW x 128],
        // tile [32 x 128], dim0 stride = KROW elems.
        v8i g1 = { (int)((1u << 16) | (1u << 20) | (3u << 22) | (3u << 25)),
                   (int)(((unsigned)KROW & 0xFFFFu) << 16),
                   (int)((((unsigned)KROW >> 16) & 0xFFFFu) | (128u << 16)),
                   (int)(32u << 16),
                   (int)128,
                   (int)KROW,
                   0, 0 };
        v4i z4 = {};
#if __clang_major__ >= 23
        v8i z8 = {};
        __builtin_amdgcn_tensor_load_to_lds(g0, g1, z4, z4, z8, 0);
#else
        __builtin_amdgcn_tensor_load_to_lds(g0, g1, z4, z4, 0);
#endif
#else
        const __bf16* bsrc = wp + (size_t)b_n * KROW + ks * 32 + b_k0;
        *(v8bf*)&Bs[buf][b_n][b_k0]     = *(const v8bf*)(bsrc);
        *(v8bf*)&Bs[buf][b_n][b_k0 + 8] = *(const v8bf*)(bsrc + 8);
#endif
    };

    v8f acc0 = {}; v8f acc1 = {};

    stageA(0, 0);
    stageB(0, 0);

    for (int ks = 0; ks < KSTEPS; ++ks) {
        const int buf = ks & 1;
        __syncthreads();                       // readers of buf^1 (ks-1) are done
        if (ks + 1 < KSTEPS) {
            stageA(ks + 1, buf ^ 1);
            stageB(ks + 1, buf ^ 1);
            WAIT_TENSOR(1);                    // TDM(ks) done; TDM(ks+1) in flight
        } else {
            WAIT_TENSOR(0);                    // drain last TDM
        }
        __syncthreads();                       // buf fully staged for everyone

        // A frag: lane row M=lrow; K = khalf*16 + e (contiguous, matches B)
        const int arow = msub * 16 + lrow;
        v8bf alo = *(const v8bf*)&As[buf][arow][khalf * 16];
        v8bf ahi = *(const v8bf*)&As[buf][arow][khalf * 16 + 8];
        v16bf afrag = __builtin_shufflevector(alo, ahi, 0,1,2,3,4,5,6,7,8,9,10,11,12,13,14,15);

        // B frag: lane col N=lrow; K = khalf*16 + e (contiguous in [N][K])
        const int bn0 = ntile0 * 16 + lrow;
        v8bf b0lo = *(const v8bf*)&Bs[buf][bn0][khalf * 16];
        v8bf b0hi = *(const v8bf*)&Bs[buf][bn0][khalf * 16 + 8];
        v16bf bfrag0 = __builtin_shufflevector(b0lo, b0hi, 0,1,2,3,4,5,6,7,8,9,10,11,12,13,14,15);
        const int bn1 = bn0 + 16;
        v8bf b1lo = *(const v8bf*)&Bs[buf][bn1][khalf * 16];
        v8bf b1hi = *(const v8bf*)&Bs[buf][bn1][khalf * 16 + 8];
        v16bf bfrag1 = __builtin_shufflevector(b1lo, b1hi, 0,1,2,3,4,5,6,7,8,9,10,11,12,13,14,15);

        acc0 = __builtin_amdgcn_wmma_f32_16x16x32_bf16(false, afrag, false, bfrag0,
                                                       (short)0, acc0, false, false);
        acc1 = __builtin_amdgcn_wmma_f32_16x16x32_bf16(false, afrag, false, bfrag1,
                                                       (short)0, acc1, false, false);
    }

    const int N0 = ntile0 * 16 + lrow;
    const int N1 = N0 + 16;
    const float bv0 = bias[N0];
    const float bv1 = bias[N1];
#pragma unroll
    for (int r = 0; r < 8; ++r) {   // lanes 0..15 share M -> 64B coalesced stores
        const int M = mbase + msub * 16 + khalf * 8 + r;
        out[(size_t)M * 128 + N0] = acc0[r] + bv0;
        out[(size_t)M * 128 + N1] = acc1[r] + bv1;
    }
}

// ------------------------------- host driver -------------------------------

extern "C" void kernel_launch(void* const* d_in, const int* in_sizes, int n_in,
                              void* d_out, int out_size, void* d_ws, size_t ws_size,
                              hipStream_t stream) {
    (void)in_sizes; (void)n_in; (void)out_size; (void)ws_size;

    // ----- input index map (setup_inputs dict insertion order, depth-first) -----
    const float* X = (const float*)d_in[0];
    auto NW1 = [&](int d) { return (const float*)d_in[1 + d * 4 + 0]; };
    auto NB1 = [&](int d) { return (const float*)d_in[1 + d * 4 + 1]; };
    auto NW2 = [&](int d) { return (const float*)d_in[1 + d * 4 + 2]; };
    auto NB2 = [&](int d) { return (const float*)d_in[1 + d * 4 + 3]; };
    auto L1W = [&](int d, int t) { return (const float*)d_in[9 + (d * 5 + t) * 2]; };
    auto L1B = [&](int d, int t) { return (const float*)d_in[10 + (d * 5 + t) * 2]; };
    auto L2W = [&](int d, int t) { return (const float*)d_in[29 + (d * 5 + t) * 2]; };
    auto L2B = [&](int d, int t) { return (const float*)d_in[30 + (d * 5 + t) * 2]; };
    auto S1W = [&](int d, int t, int hc) { return (const float*)d_in[49 + ((d * 4 + (t - 1)) * 2 + hc) * 2]; };
    auto S1B = [&](int d, int t, int hc) { return (const float*)d_in[50 + ((d * 4 + (t - 1)) * 2 + hc) * 2]; };
    auto S2W = [&](int t, int hc) { return (const float*)d_in[81 + (t * 2 + hc) * 2]; };
    auto S2B = [&](int t, int hc) { return (const float*)d_in[82 + (t * 2 + hc) * 2]; };

    // ----- workspace carve-up -----
    char* ws = (char*)d_ws;
    size_t off = 0;
    auto take = [&](size_t bytes) -> char* {
        char* p = ws + off; off = (off + bytes + 255) & ~(size_t)255; return p;
    };
    float* xsl    = (float*)take(sizeof(float) * (size_t)TT * SLICE);  // head / initial cur
    float* curA   = (float*)take(sizeof(float) * (size_t)TT * SLICE);
    float* curB   = (float*)take(sizeof(float) * (size_t)TT * SLICE);
    float* pastH  = (float*)take(sizeof(float) * (size_t)TT * SLICE);  // slot t = h_{t+1}
    float* pastC  = (float*)take(sizeof(float) * (size_t)TT * SLICE);
    float* deepH  = (float*)take(sizeof(float) * (size_t)TT * SLICE);
    float* deepC  = (float*)take(sizeof(float) * (size_t)TT * SLICE);
    float* selCur = (float*)take(sizeof(float) * SLICE);
    float* selS   = (float*)take(sizeof(float) * SLICE);
    float* selH   = (float*)take(sizeof(float) * SLICE);
    float* selC   = (float*)take(sizeof(float) * SLICE);
    float* selMH  = (float*)take(sizeof(float) * SLICE);
    float* selMC  = (float*)take(sizeof(float) * SLICE);
    float* zerob  = (float*)take(sizeof(float) * SLICE);
    float* ccb    = (float*)take(sizeof(float) * (size_t)4 * SLICE);
    float* mmb    = (float*)take(sizeof(float) * (size_t)4 * SLICE);
    __bf16* pk1   = (__bf16*)take(sizeof(__bf16) * (size_t)NPIX * 64);
    __bf16* pk2   = (__bf16*)take(sizeof(__bf16) * (size_t)NPIX * 128);
    __bf16* wp1[2], *wp2[2];
    for (int d = 0; d < 2; ++d) {
        wp1[d] = (__bf16*)take(sizeof(__bf16) * 128 * 576);
        wp2[d] = (__bf16*)take(sizeof(__bf16) * 128 * 1152);
    }

    auto G = [](int n) { return dim3((n + 255) / 256); };

    // ----- prologue -----
    k_transpose_in<<<G(TT * SLICE), 256, 0, stream>>>(X, xsl, TT * SLICE);
    k_zero<<<G(SLICE), 256, 0, stream>>>(zerob, SLICE);
    for (int d = 0; d < 2; ++d) {
        k_wpack<<<G(128 * 576), 256, 0, stream>>>(NW1(d), wp1[d], 64, 128 * 576);
        k_wpack<<<G(128 * 1152), 256, 0, stream>>>(NW2(d), wp2[d], 128, 128 * 1152);
    }

    // ----- recurrent chain -----
    for (int d = 0; d < 2; ++d) {
        const float* curSrc = (d == 0) ? xsl : curA;
        float* curDst = (d == 0) ? curA : curB;
        for (int t = 0; t < TT; ++t) {
            k_mix_select<<<G(SLICE), 256, 0, stream>>>(selCur, curSrc, L1W(d, t), L1B(d, t),
                                                       5, 0, (long long)SLICE, SLICE);
            k_mix_select<<<G(SLICE), 256, 0, stream>>>(selS, xsl, L2W(d, t), L2B(d, t),
                                                       5, 0, (long long)SLICE, SLICE);
            const float* sh = zerob; const float* sc = zerob;
            if (t > 0) {
                k_mix_select<<<G(SLICE), 256, 0, stream>>>(selH, pastH, S1W(d, t, 0), S1B(d, t, 0),
                                                           t + 1, 1, (long long)SLICE, SLICE);
                k_mix_select<<<G(SLICE), 256, 0, stream>>>(selC, pastC, S1W(d, t, 1), S1B(d, t, 1),
                                                           t + 1, 1, (long long)SLICE, SLICE);
                sh = selH; sc = selC;
            }
            const float* smh = zerob; const float* smc = zerob;
            if (d > 0) {   // stack = [zeros, deep_{d-1}[t]]
                k_mix_select<<<G(SLICE), 256, 0, stream>>>(selMH, deepH + (size_t)t * SLICE,
                                                           S2W(t, 0), S2B(t, 0), 2, 1, 0, SLICE);
                k_mix_select<<<G(SLICE), 256, 0, stream>>>(selMC, deepC + (size_t)t * SLICE,
                                                           S2W(t, 1), S2B(t, 1), 2, 1, 0, SLICE);
                smh = selMH; smc = selMC;
            }
            k_pack2<<<G(NPIX * 64), 256, 0, stream>>>(pk1, selCur, sh, NPIX * 64);
            k_conv<64><<<NPIX / 32, 256, 0, stream>>>(pk1, wp1[d], NB1(d), ccb);
            k_pack4<<<G(NPIX * 128), 256, 0, stream>>>(pk2, selCur, sh, selS, smh, NPIX * 128);
            k_conv<128><<<NPIX / 32, 256, 0, stream>>>(pk2, wp2[d], NB2(d), mmb);
            k_gates<<<G(SLICE), 256, 0, stream>>>(ccb, mmb, sc, smc,
                                                  pastH + (size_t)t * SLICE,
                                                  pastC + (size_t)t * SLICE,
                                                  deepH + (size_t)t * SLICE,
                                                  deepC + (size_t)t * SLICE,
                                                  curDst + (size_t)t * SLICE, SLICE);
        }
    }

    k_unpack<<<G(TT * SLICE), 256, 0, stream>>>(curB, (float*)d_out, TT * SLICE);
}
